// EGATClassifier_38809324486857
// MI455X (gfx1250) — matrix operand order repaired and checked by reading, not compiled, not run on hardware
//
#include <hip/hip_runtime.h>

typedef __bf16 bf16;
typedef __attribute__((ext_vector_type(16))) __bf16        v16bf;
typedef __attribute__((ext_vector_type(8)))  float         v8f;
typedef __attribute__((ext_vector_type(4)))  unsigned int  v4u;

#define GN   50000
#define GE   800000
#define GD   64
#define GH   4
#define GHD  256
#define GNC  2
#define GB   8
#define GCL  2

// ---------------------------------------------------------------- helpers
__device__ __forceinline__ void atomicMaxF(float* addr, float val) {
    if (val >= 0.0f) atomicMax((int*)addr, __float_as_int(val));
    else             atomicMin((unsigned int*)addr, (unsigned int)__float_as_int(val));
}

// ---------------------------------------------------------------- elementwise
__global__ void k_cvt_f32_bf16(const float* __restrict__ in, bf16* __restrict__ out, long long n) {
    long long i  = (long long)blockIdx.x * blockDim.x + threadIdx.x;
    long long st = (long long)gridDim.x * blockDim.x;
    for (; i < n; i += st) out[i] = (bf16)in[i];
}

__global__ void k_fill(float* __restrict__ p, float v, long long n) {
    long long i  = (long long)blockIdx.x * blockDim.x + threadIdx.x;
    long long st = (long long)gridDim.x * blockDim.x;
    for (; i < n; i += st) p[i] = v;
}

// W (64 x 256) f32  ->  Wt (256 x 64) bf16
__global__ void k_transpose_cvt(const float* __restrict__ in, bf16* __restrict__ out) {
    int i = blockIdx.x * blockDim.x + threadIdx.x;   // over 64*256
    if (i >= GD * GHD) return;
    int k = i / GHD, c = i % GHD;
    out[c * GD + k] = (bf16)in[i];
}

// ---------------------------------------------------------------- node GEMM
// out[item][col] (M x 256 bf16) = rows(M x 64 bf16) x Wt(256 x 64 bf16)^T
// WMMA mapping: A = Wt tile (cols=M dim), B = 16 items (N dim), K=64 in 2 steps.
__global__ __launch_bounds__(256) void k_gemm_rows_wt(
    const bf16* __restrict__ rows, const bf16* __restrict__ wt,
    bf16* __restrict__ out, int M)
{
    __shared__ __align__(16) bf16 sWt[GHD * GD];          // 32 KB
    {
        const v4u* g = (const v4u*)wt;
        v4u*       l = (v4u*)sWt;
        for (int j = threadIdx.x; j < (GHD * GD) / 8; j += 256) l[j] = g[j];
    }
    __syncthreads();

    const int wave = threadIdx.x >> 5;
    const int lane = threadIdx.x & 31;
    const int tile = blockIdx.x * 8 + wave;
    if (tile * 16 >= M) return;

    const int item = tile * 16 + (lane & 15);
    const int hi8  = (lane >> 4) * 8;

    // B fragment (32x16 bf16): lane n=item, K contiguous: lanes>=16 hold K+16
    const bf16* rp = rows + (size_t)item * GD + (lane >> 4) * 16;
    union { v16bf v; v4u u[2]; } bfrag[2];
    bfrag[0].u[0] = *(const v4u*)(rp);
    bfrag[0].u[1] = *(const v4u*)(rp + 8);
    bfrag[1].u[0] = *(const v4u*)(rp + 32);
    bfrag[1].u[1] = *(const v4u*)(rp + 40);

    #pragma unroll
    for (int mt = 0; mt < 16; ++mt) {
        v8f acc = {};
        #pragma unroll
        for (int s = 0; s < 2; ++s) {
            // A fragment (16x32 bf16): lane m = lane&15; K contiguous: lanes>=16 hold K+16
            const bf16* ap = sWt + (size_t)(mt * 16 + (lane & 15)) * GD + s * 32 + (lane >> 4) * 16;
            union { v16bf v; v4u u[2]; } afrag;
            afrag.u[0] = *(const v4u*)(ap);
            afrag.u[1] = *(const v4u*)(ap + 8);
            acc = __builtin_amdgcn_wmma_f32_16x16x32_bf16(
                false, afrag.v, false, bfrag[s].v, (short)0, acc, false, false);
        }
        union { v4u u; bf16 h[8]; } o;
        #pragma unroll
        for (int v = 0; v < 8; ++v) o.h[v] = (bf16)acc[v];
        *(v4u*)(out + (size_t)item * GHD + mt * 16 + hi8) = o.u;
    }
}

// ---------------------------------------------------------------- fused edge kernel
// Per 16-edge tile: WMMA  w@Wfij, add gathered f_ni[src], f_nj[dst], bias,
// LeakyReLU, attention logits a (E x 4), head-mean w_next (in place over w).
__global__ __launch_bounds__(256) void k_edge_fused(
    const bf16* __restrict__ wrows, const bf16* __restrict__ wt,
    const bf16* __restrict__ f_ni,  const bf16* __restrict__ f_nj,
    const int* __restrict__ src,    const int* __restrict__ dst,
    const float* __restrict__ bias, const float* __restrict__ attn,
    float* __restrict__ a_out,      bf16* __restrict__ w_out)
{
    __shared__ __align__(16) bf16 sWt[GHD * GD];
    {
        const v4u* g = (const v4u*)wt;
        v4u*       l = (v4u*)sWt;
        for (int j = threadIdx.x; j < (GHD * GD) / 8; j += 256) l[j] = g[j];
    }
    __syncthreads();

    const int wave = threadIdx.x >> 5;
    const int lane = threadIdx.x & 31;
    const int tile = blockIdx.x * 8 + wave;
    if (tile * 16 >= GE) return;

    const int e   = tile * 16 + (lane & 15);
    const int hi8 = (lane >> 4) * 8;
    const int sr  = src[e];
    const int dr  = dst[e];

    const bf16* rp = wrows + (size_t)e * GD + (lane >> 4) * 16;
    union { v16bf v; v4u u[2]; } bfrag[2];
    bfrag[0].u[0] = *(const v4u*)(rp);
    bfrag[0].u[1] = *(const v4u*)(rp + 8);
    bfrag[1].u[0] = *(const v4u*)(rp + 32);
    bfrag[1].u[1] = *(const v4u*)(rp + 40);

    float wsum[32];
    #pragma unroll
    for (int i = 0; i < 32; ++i) wsum[i] = 0.0f;
    float alog[4] = {0.f, 0.f, 0.f, 0.f};

    #pragma unroll
    for (int mt = 0; mt < 16; ++mt) {
        v8f acc = {};
        #pragma unroll
        for (int s = 0; s < 2; ++s) {
            const bf16* ap = sWt + (size_t)(mt * 16 + (lane & 15)) * GD + s * 32 + (lane >> 4) * 16;
            union { v16bf v; v4u u[2]; } afrag;
            afrag.u[0] = *(const v4u*)(ap);
            afrag.u[1] = *(const v4u*)(ap + 8);
            acc = __builtin_amdgcn_wmma_f32_16x16x32_bf16(
                false, afrag.v, false, bfrag[s].v, (short)0, acc, false, false);
        }
        const int head = mt >> 2;
        const int r    = mt & 3;
        const int c0   = mt * 16 + hi8;
        union { v4u u; bf16 h[8]; } niu, nju;
        niu.u = *(const v4u*)(f_ni + (size_t)sr * GHD + c0);
        nju.u = *(const v4u*)(f_nj + (size_t)dr * GHD + c0);
        float part = 0.0f;
        #pragma unroll
        for (int v = 0; v < 8; ++v) {
            float ev = acc[v] + (float)niu.h[v] + (float)nju.h[v] + bias[c0 + v];
            ev = ev > 0.0f ? ev : 0.01f * ev;                       // LeakyReLU
            part += ev * attn[head * GD + r * 16 + hi8 + v];
            wsum[r * 8 + v] += ev;
        }
        alog[head] += part;
    }

    // a[e][h] = sum over both lane halves (cols 0-7 and 8-15 of each 16-tile)
    #pragma unroll
    for (int h = 0; h < 4; ++h) {
        float o = alog[h] + __shfl_xor(alog[h], 16, 32);
        if (lane < 16) a_out[(size_t)e * GH + h] = o;
    }
    // w_next[e][d] = mean over heads of e  (in-place safe: tile rows fully read)
    #pragma unroll
    for (int r = 0; r < 4; ++r) {
        union { v4u u; bf16 h[8]; } o;
        #pragma unroll
        for (int v = 0; v < 8; ++v) o.h[v] = (bf16)(0.25f * wsum[r * 8 + v]);
        *(v4u*)(w_out + (size_t)e * GD + r * 16 + hi8) = o.u;
    }
}

// ---------------------------------------------------------------- softmax + scatter
__global__ void k_seg_amax(const float* __restrict__ a, const int* __restrict__ dst,
                           float* __restrict__ amax) {
    long long i  = (long long)blockIdx.x * blockDim.x + threadIdx.x;
    long long st = (long long)gridDim.x * blockDim.x;
    for (; i < (long long)GE * GH; i += st) {
        int e = (int)(i >> 2), h = (int)(i & 3);
        atomicMaxF(&amax[(size_t)dst[e] * GH + h], a[i]);
    }
}

__global__ void k_exp_sum(float* __restrict__ a, const int* __restrict__ dst,
                          const float* __restrict__ amax, float* __restrict__ denom) {
    long long i  = (long long)blockIdx.x * blockDim.x + threadIdx.x;
    long long st = (long long)gridDim.x * blockDim.x;
    for (; i < (long long)GE * GH; i += st) {
        int e = (int)(i >> 2), h = (int)(i & 3);
        int d = dst[e];
        float ex = __expf(a[i] - amax[(size_t)d * GH + h]);
        a[i] = ex;
        atomicAdd(&denom[(size_t)d * GH + h], ex);
    }
}

// h_next[dst][d] += (1/H) * sum_h alpha[e,h] * h_proj[src][h*64+d]
__global__ void k_scatter(const float* __restrict__ ex, const float* __restrict__ denom,
                          const int* __restrict__ src, const int* __restrict__ dst,
                          const bf16* __restrict__ h_proj, float* __restrict__ hnext) {
    long long i  = (long long)blockIdx.x * blockDim.x + threadIdx.x;
    long long st = (long long)gridDim.x * blockDim.x;
    for (; i < (long long)GE * GD; i += st) {
        int e = (int)(i >> 6), d = (int)(i & 63);
        int sr = src[e], dr = dst[e];
        float s = 0.0f;
        #pragma unroll
        for (int h = 0; h < GH; ++h) {
            float alpha = ex[(size_t)e * GH + h] / denom[(size_t)dr * GH + h];
            s += alpha * (float)h_proj[(size_t)sr * GHD + h * GD + d];
        }
        atomicAdd(&hnext[(size_t)dr * GD + d], 0.25f * s);
    }
}

// ---------------------------------------------------------------- JK concat + pool + linear
__global__ __launch_bounds__(256) void k_classifier(
    const float* __restrict__ feat, const float* __restrict__ hs,
    const float* __restrict__ W_lin, const float* __restrict__ b_lin,
    float* __restrict__ out)
{
    __shared__ float pool[(GNC + 1) * GD];
    const int b = blockIdx.x, t = threadIdx.x;
    const int npg = GN / GB;
    if (t < (GNC + 1) * GD) {
        const float* sp; int off;
        if (t < GD)          { sp = feat;                      off = t;        }
        else if (t < 2 * GD) { sp = hs;                        off = t - GD;   }
        else                 { sp = hs + (size_t)GN * GD;      off = t - 2*GD; }
        float s = 0.0f;
        for (int n = b * npg; n < (b + 1) * npg; ++n) s += sp[(size_t)n * GD + off];
        pool[t] = s / (float)npg;
    }
    __syncthreads();
    if (t < GCL) {
        float s = b_lin[t];
        #pragma unroll 4
        for (int j = 0; j < (GNC + 1) * GD; ++j) s += pool[j] * W_lin[j * GCL + t];
        out[b * GCL + t] = s;
    }
}

// ---------------------------------------------------------------- launch
extern "C" void kernel_launch(void* const* d_in, const int* in_sizes, int n_in,
                              void* d_out, int out_size, void* d_ws, size_t ws_size,
                              hipStream_t stream) {
    const float* feat    = (const float*)d_in[0];
    const float* eweight = (const float*)d_in[1];
    const int*   src     = (const int*)  d_in[2];
    const int*   dst     = (const int*)  d_in[3];
    const float* W_node  = (const float*)d_in[4];
    const float* W_ni    = (const float*)d_in[5];
    const float* W_nj    = (const float*)d_in[6];
    const float* W_fij   = (const float*)d_in[7];
    const float* attn    = (const float*)d_in[8];
    const float* bias_e  = (const float*)d_in[9];
    const float* W_lin   = (const float*)d_in[10];
    const float* b_lin   = (const float*)d_in[11];

    // ---- workspace carve (all regions written before read)
    char* w = (char*)d_ws;
    auto take = [&](size_t bytes) { void* p = w; w += (bytes + 255) & ~(size_t)255; return p; };
    bf16*  h_bf   = (bf16*) take((size_t)GN * GD * 2);
    bf16*  w_bf   = (bf16*) take((size_t)GE * GD * 2);          // also w_next (in place)
    bf16*  Wt     = (bf16*) take((size_t)GNC * 4 * GHD * GD * 2);
    bf16*  f_ni   = (bf16*) take((size_t)GN * GHD * 2);
    bf16*  f_nj   = (bf16*) take((size_t)GN * GHD * 2);
    bf16*  h_proj = (bf16*) take((size_t)GN * GHD * 2);
    float* aL     = (float*)take((size_t)GE * GH * 4);
    float* amax   = (float*)take((size_t)GN * GH * 4);
    float* denom  = (float*)take((size_t)GN * GH * 4);
    float* hs     = (float*)take((size_t)GNC * GN * GD * 4);

    const int T = 256;
    auto gsz = [&](long long n) { long long b = (n + T - 1) / T; return (int)(b > 1048576 ? 1048576 : b); };

    // bf16 copies of activations
    k_cvt_f32_bf16<<<gsz((long long)GN * GD), T, 0, stream>>>(feat,    h_bf, (long long)GN * GD);
    k_cvt_f32_bf16<<<gsz((long long)GE * GD), T, 0, stream>>>(eweight, w_bf, (long long)GE * GD);

    // transposed bf16 weights: [layer][0=ni,1=nj,2=fij,3=node]
    const float* wsrc[4] = {W_ni, W_nj, W_fij, W_node};
    for (int i = 0; i < GNC; ++i)
        for (int m = 0; m < 4; ++m)
            k_transpose_cvt<<<(GD * GHD + T - 1) / T, T, 0, stream>>>(
                wsrc[m] + (size_t)i * GD * GHD, Wt + ((size_t)i * 4 + m) * GHD * GD);

    const int nodeBlocks = ((GN / 16) + 7) / 8;   // 3125 wave-tiles
    const int edgeBlocks = (GE / 16) / 8;         // 50000 wave-tiles exactly

    for (int i = 0; i < GNC; ++i) {
        const bf16* WtL = Wt + (size_t)i * 4 * GHD * GD;
        // node projections (bf16 outputs -> L2-resident gather tables)
        k_gemm_rows_wt<<<nodeBlocks, T, 0, stream>>>(h_bf, WtL + 0 * GHD * GD, f_ni,   GN);
        k_gemm_rows_wt<<<nodeBlocks, T, 0, stream>>>(h_bf, WtL + 1 * GHD * GD, f_nj,   GN);
        k_gemm_rows_wt<<<nodeBlocks, T, 0, stream>>>(h_bf, WtL + 3 * GHD * GD, h_proj, GN);

        float* hnext = hs + (size_t)i * GN * GD;
        k_fill<<<gsz((long long)GN * GH), T, 0, stream>>>(amax, -INFINITY, (long long)GN * GH);
        k_fill<<<gsz((long long)GN * GH), T, 0, stream>>>(denom, 0.0f,     (long long)GN * GH);
        k_fill<<<gsz((long long)GN * GD), T, 0, stream>>>(hnext, 0.0f,     (long long)GN * GD);

        // fused edge GEMM + gather + LeakyReLU + attention logits + head-mean w update
        k_edge_fused<<<edgeBlocks, T, 0, stream>>>(
            w_bf, WtL + 2 * GHD * GD, f_ni, f_nj, src, dst,
            bias_e + (size_t)i * GHD, attn + (size_t)i * GH * GD, aL, w_bf);

        // edge softmax over incoming edges + message scatter (head-mean fused)
        k_seg_amax<<<gsz((long long)GE * GH), T, 0, stream>>>(aL, dst, amax);
        k_exp_sum <<<gsz((long long)GE * GH), T, 0, stream>>>(aL, dst, amax, denom);
        k_scatter <<<gsz((long long)GE * GD), T, 0, stream>>>(aL, denom, src, dst, h_proj, hnext);

        // next-layer bf16 activations
        k_cvt_f32_bf16<<<gsz((long long)GN * GD), T, 0, stream>>>(hnext, h_bf, (long long)GN * GD);
    }

    k_classifier<<<GB, T, 0, stream>>>(feat, hs, W_lin, b_lin, (float*)d_out);
}
